// Multihead_57750130262132
// MI455X (gfx1250) — matrix operand order repaired and checked by reading, not compiled, not run on hardware
//
#include <hip/hip_runtime.h>

// ---------------------------------------------------------------------------
// Causal multi-head attention forward for MI455X (gfx1250, wave32, WMMA).
//   1) convert x -> bf16; transpose-convert Wq/Wk/Wv -> [H][D][E] bf16,
//      Wc -> [E][H*D] bf16  (B-operand of WMMA wants column-contiguous data)
//   2) QKV GEMM (software-pipelined): Q,K row-major [b,h,t,d]; V -> [b,h,d,t]
//   3) flash-attention: 8 waves/block share one (b,h); K/V tiles DMA'd into
//      double-buffered block LDS by the Tensor Data Mover (tensor_load_to_lds,
//      TENSORcnt), one step ahead of compute; online softmax; P staged through
//      wave-private LDS with s_wait_dscnt ordering
//   4) output GEMM (software-pipelined): O[B*T,1024] @ Wc -> fp32 d_out
// ---------------------------------------------------------------------------

constexpr int Bc = 4, Tc = 2048, Ec = 1024, Hc = 16, Dc = 64;

typedef __attribute__((ext_vector_type(16))) __bf16       v16bf;
typedef __attribute__((ext_vector_type(8)))  float        v8f;
typedef __attribute__((ext_vector_type(4)))  unsigned int v4u;
typedef __attribute__((ext_vector_type(8)))  int          v8i;
typedef __attribute__((ext_vector_type(4)))  int          v4i;

union FragU { v4u h[2]; v16bf bf; };

static __device__ __forceinline__ unsigned short bf16r(float f) {
  unsigned int u = __builtin_bit_cast(unsigned int, f);
  u += 0x7FFFu + ((u >> 16) & 1u);           // round-to-nearest-even
  return (unsigned short)(u >> 16);
}

// A fragment, 16x32 bf16 (M x K), row-major memory, rows = lane&15.
// ISA layout: lanes 0-15 hold K = {k0..k0+7, k0+16..k0+23}; lanes 16-31 +8.
static __device__ __forceinline__ v16bf load_fragA(const unsigned short* base,
                                                   int stride, int k0, int lane) {
  int r = lane & 15, hi = lane >> 4;
  const unsigned short* p = base + (size_t)r * stride + k0 + hi * 8;
  FragU u;
  u.h[0] = *(const v4u*)p;
  u.h[1] = *(const v4u*)(p + 16);
  return u.bf;
}

// B fragment, 32x16 bf16 (K x N); memory holds B^T row-major (col-contig).
// ISA layout: lanes 0-15 hold K = {k0..k0+7, k0+16..k0+23}; lanes 16-31 +8.
static __device__ __forceinline__ v16bf load_fragB(const unsigned short* base,
                                                   int stride, int k0, int lane) {
  int r = lane & 15, hi = lane >> 4;
  const unsigned short* p = base + (size_t)r * stride + k0 + hi * 8;
  FragU u;
  u.h[0] = *(const v4u*)p;
  u.h[1] = *(const v4u*)(p + 16);
  return u.bf;
}

static __device__ __forceinline__ v8f wmma_bf16(v16bf a, v16bf b, v8f c) {
  return __builtin_amdgcn_wmma_f32_16x16x32_bf16(false, a, false, b,
                                                 (short)0, c, false, false);
}

// Tensor Data Mover: DMA a 2D tile (rows x rowlen bf16 elems, row stride in
// elems) from global into LDS. D# packed per CDNA5 ISA ch.8 (groups 0/1;
// groups 2/3 zero for 2D). Completion tracked with TENSORcnt.
// This toolchain exposes the 6-arg builtin: (v4u, v8i, v4i, v4i, v8i, cpol).
static __device__ __forceinline__ void tdm_load_2d(const unsigned short* gptr,
                                                   unsigned short* lptr,
                                                   unsigned rows, unsigned rowlen,
                                                   unsigned stride_elems) {
  unsigned long long ga = (unsigned long long)(uintptr_t)gptr;
  unsigned lds = (unsigned)(uintptr_t)lptr;      // generic->LDS: low 32 bits
  v4u g0;
  g0[0] = 1u;                                    // count=1, user descriptor
  g0[1] = lds;                                   // lds_addr (bytes)
  g0[2] = (unsigned)(ga & 0xFFFFFFFFu);          // global_addr[31:0]
  g0[3] = (unsigned)((ga >> 32) & 0x01FFFFFFu) | (2u << 30);  // addr hi | type=2
  v8i g1;
  g1[0] = (int)(1u << 16);                             // data_size=1 (2 bytes)
  g1[1] = (int)((rowlen & 0xFFFFu) << 16);             // tensor_dim0 lo16
  g1[2] = (int)((rowlen >> 16) | ((rows & 0xFFFFu) << 16)); // dim0 hi / dim1 lo
  g1[3] = (int)((rows >> 16) | ((rowlen & 0xFFFFu) << 16)); // dim1 hi / tile_dim0
  g1[4] = (int)(rows & 0xFFFFu);                       // tile_dim1=rows, tile_dim2=0
  g1[5] = (int)stride_elems;                           // tensor_dim0_stride lo32
  g1[6] = 0;                                           // stride hi16 / dim1_stride lo
  g1[7] = 0;
  v4i gz4 = {0, 0, 0, 0};
  v8i gz8 = {0, 0, 0, 0, 0, 0, 0, 0};
  __builtin_amdgcn_tensor_load_to_lds(g0, g1, gz4, gz4, gz8, 0);
}

// ------------------------------ prep kernels -------------------------------

__global__ void __launch_bounds__(256) cvt_bf16(const float* __restrict__ in,
                                                unsigned short* __restrict__ out,
                                                long n) {
  for (long i = (long)blockIdx.x * blockDim.x + threadIdx.x; i < n;
       i += (long)gridDim.x * blockDim.x)
    out[i] = bf16r(in[i]);
}

// in[b][r][c] (f32) -> out[b][c][r] (bf16)
__global__ void __launch_bounds__(256) transpose_bf16(const float* __restrict__ in,
                                                      unsigned short* __restrict__ out,
                                                      int batch, int R, int C) {
  long n = (long)batch * R * C;
  for (long i = (long)blockIdx.x * blockDim.x + threadIdx.x; i < n;
       i += (long)gridDim.x * blockDim.x) {
    long b = i / ((long)R * C);
    long rem = i - b * (long)R * C;
    long c = rem / R;
    long r = rem - c * R;
    out[i] = bf16r(in[(b * R + r) * C + c]);
  }
}

// ------------------------------- QKV GEMM ----------------------------------
// One wave: 16 rows of x times one head's W (E=1024 k-loop), 16x64 output.
// Software-pipelined: next k-step fragments prefetched during current WMMAs.
__global__ void __launch_bounds__(256) qkv_gemm(
    const unsigned short* __restrict__ xb,
    const unsigned short* __restrict__ WqT, const unsigned short* __restrict__ WkT,
    const unsigned short* __restrict__ WvT,
    unsigned short* __restrict__ Q, unsigned short* __restrict__ K,
    unsigned short* __restrict__ Vt) {
  __shared__ unsigned short stg[8][16 * 64];
  int wave = threadIdx.x >> 5, lane = threadIdx.x & 31;
  int wid = blockIdx.x * 8 + wave;
  int sblk = wid & 127;
  int t = wid >> 7;
  int h = t & 15; t >>= 4;
  int b = t & 3;
  int mat = t >> 2;                                  // 0=Q 1=K 2=V
  const unsigned short* WT = (mat == 0) ? WqT : (mat == 1) ? WkT : WvT;
  const unsigned short* Ab = xb + ((size_t)b * Tc + sblk * 16) * Ec;
  const unsigned short* Bb = WT + (size_t)h * Dc * Ec;

  v8f z = {0.f, 0.f, 0.f, 0.f, 0.f, 0.f, 0.f, 0.f};
  v8f acc[4];
#pragma unroll
  for (int i = 0; i < 4; i++) acc[i] = z;

  v16bf a  = load_fragA(Ab, Ec, 0, lane);
  v16bf b0 = load_fragB(Bb + 0 * 16 * Ec, Ec, 0, lane);
  v16bf b1 = load_fragB(Bb + 1 * 16 * Ec, Ec, 0, lane);
  v16bf b2 = load_fragB(Bb + 2 * 16 * Ec, Ec, 0, lane);
  v16bf b3 = load_fragB(Bb + 3 * 16 * Ec, Ec, 0, lane);
  for (int e0 = 0; e0 < Ec - 32; e0 += 32) {
    v16bf an = load_fragA(Ab, Ec, e0 + 32, lane);
    v16bf c0 = load_fragB(Bb + 0 * 16 * Ec, Ec, e0 + 32, lane);
    v16bf c1 = load_fragB(Bb + 1 * 16 * Ec, Ec, e0 + 32, lane);
    v16bf c2 = load_fragB(Bb + 2 * 16 * Ec, Ec, e0 + 32, lane);
    v16bf c3 = load_fragB(Bb + 3 * 16 * Ec, Ec, e0 + 32, lane);
    acc[0] = wmma_bf16(a, b0, acc[0]);
    acc[1] = wmma_bf16(a, b1, acc[1]);
    acc[2] = wmma_bf16(a, b2, acc[2]);
    acc[3] = wmma_bf16(a, b3, acc[3]);
    a = an; b0 = c0; b1 = c1; b2 = c2; b3 = c3;
  }
  acc[0] = wmma_bf16(a, b0, acc[0]);
  acc[1] = wmma_bf16(a, b1, acc[1]);
  acc[2] = wmma_bf16(a, b2, acc[2]);
  acc[3] = wmma_bf16(a, b3, acc[3]);

  int n = lane & 15, hi = lane >> 4;
  if (mat < 2) {
    // stage D-layout -> row-major [s][d] via LDS, then 16B/lane stores
    unsigned short* tt = &stg[wave][0];
#pragma unroll
    for (int nt = 0; nt < 4; nt++)
#pragma unroll
      for (int j = 0; j < 8; j++)
        tt[(j + 8 * hi) * 64 + nt * 16 + n] = bf16r(acc[nt][j]);
    asm volatile("s_wait_dscnt 0" ::: "memory");
    unsigned short* out = (mat == 0 ? Q : K) +
        ((size_t)(b * Hc + h) * Tc + sblk * 16 + n) * Dc + hi * 32;
    const unsigned short* src = tt + n * 64 + hi * 32;
#pragma unroll
    for (int c = 0; c < 4; c++) ((v4u*)out)[c] = ((const v4u*)src)[c];
  } else {
    // V transposed store: D-layout row index (t) is contiguous per lane
#pragma unroll
    for (int nt = 0; nt < 4; nt++) {
      union { unsigned short s[8]; v4u v; } pk;
#pragma unroll
      for (int j = 0; j < 8; j++) pk.s[j] = bf16r(acc[nt][j]);
      int d = nt * 16 + n;
      *(v4u*)(Vt + ((size_t)(b * Hc + h) * Dc + d) * Tc + sblk * 16 + hi * 8) = pk.v;
    }
  }
}

// --------------------------- flash attention -------------------------------
// 8 waves/block share one (b,h); wave w owns query rows [qs, qs+16).
// K/V 32-key tiles are TDM-DMA'd into double-buffered block LDS one step
// ahead (wave 0 issues, TENSORcnt ordering, barriers publish to all waves).
__global__ void __launch_bounds__(256) attn_kernel(
    const unsigned short* __restrict__ Q, const unsigned short* __restrict__ K,
    const unsigned short* __restrict__ Vt, unsigned short* __restrict__ O) {
  __shared__ unsigned short pst[8][16 * 32];    // P staging (per wave)
  __shared__ unsigned short ost[8][16 * 64];    // O store staging (per wave)
  __shared__ unsigned short kbuf[2][32 * 64];   // K tile [t][d], double buffer
  __shared__ unsigned short vbuf[2][64 * 32];   // V tile [d][t], double buffer
  int wave = threadIdx.x >> 5, lane = threadIdx.x & 31;
  int wid = blockIdx.x * 8 + wave;
  int qblk = wid & 127;
  int bh = wid >> 7;                            // b*H + h (same for all waves)
  int qs = qblk * 16;
  int n = lane & 15, hi = lane >> 4;

  const unsigned short* Qb = Q + ((size_t)bh * Tc + qs) * Dc;
  const unsigned short* Kb = K + (size_t)bh * Tc * Dc;
  const unsigned short* Vb = Vt + (size_t)bh * Dc * Tc;

  v16bf qa0 = load_fragA(Qb, Dc, 0, lane);
  v16bf qa1 = load_fragA(Qb, Dc, 32, lane);

  v8f z = {0.f, 0.f, 0.f, 0.f, 0.f, 0.f, 0.f, 0.f};
  v8f o[4];
#pragma unroll
  for (int i = 0; i < 4; i++) o[i] = z;
  float mrow[8], lrow[8];
#pragma unroll
  for (int j = 0; j < 8; j++) { mrow[j] = -__builtin_inff(); lrow[j] = 0.f; }

  unsigned short* pb = &pst[wave][0];
  const float scale = 0.125f;                    // 1/sqrt(64)

  // uniform trip count across the block (max causal extent of its 8 waves)
  int base_qblk = (blockIdx.x * 8) & 127;
  int nsteps = (base_qblk * 16 + 128) / 32;

  if (wave == 0) {                               // preload step 0 into buf 0
    tdm_load_2d(Kb, &kbuf[0][0], 32, 64, 64);
    tdm_load_2d(Vb, &vbuf[0][0], 64, 32, Tc);
  }

  for (int s = 0; s < nsteps; ++s) {
    int tk = s * 32;
    int cur = s & 1;
    if (wave == 0) {
      if (s + 1 < nsteps) {                      // prefetch next step
        int tn = tk + 32;
        tdm_load_2d(Kb + (size_t)tn * Dc, &kbuf[cur ^ 1][0], 32, 64, 64);
        tdm_load_2d(Vb + tn, &vbuf[cur ^ 1][0], 64, 32, Tc);
        __builtin_amdgcn_s_wait_tensorcnt(2);    // current pair complete
      } else {
        __builtin_amdgcn_s_wait_tensorcnt(0);
      }
    }
    __syncthreads();                             // publish buf[cur]

    if (tk < qs + 16) {                          // causal: real work only
      const unsigned short* kb = &kbuf[cur][0];
      const unsigned short* vb = &vbuf[cur][0];
      v8f s0 = z, s1 = z;
      s0 = wmma_bf16(qa0, load_fragB(kb, 64, 0, lane), s0);
      s0 = wmma_bf16(qa1, load_fragB(kb, 64, 32, lane), s0);
      s1 = wmma_bf16(qa0, load_fragB(kb + 16 * 64, 64, 0, lane), s1);
      s1 = wmma_bf16(qa1, load_fragB(kb + 16 * 64, 64, 32, lane), s1);

      asm volatile("s_wait_dscnt 0" ::: "memory");  // WAR on pb reload
#pragma unroll
      for (int j = 0; j < 8; j++) {
        int row = qs + j + 8 * hi;
        float a = s0[j] * scale; if (tk + n > row)      a = -__builtin_inff();
        float c = s1[j] * scale; if (tk + 16 + n > row) c = -__builtin_inff();
        float mx = fmaxf(a, c);
        mx = fmaxf(mx, __shfl_xor(mx, 1, 32));
        mx = fmaxf(mx, __shfl_xor(mx, 2, 32));
        mx = fmaxf(mx, __shfl_xor(mx, 4, 32));
        mx = fmaxf(mx, __shfl_xor(mx, 8, 32));
        float mnew  = fmaxf(mrow[j], mx);
        float alpha = __expf(mrow[j] - mnew);
        float pa = __expf(a - mnew);
        float pc = __expf(c - mnew);
        float sm = pa + pc;
        sm += __shfl_xor(sm, 1, 32);
        sm += __shfl_xor(sm, 2, 32);
        sm += __shfl_xor(sm, 4, 32);
        sm += __shfl_xor(sm, 8, 32);
        lrow[j] = lrow[j] * alpha + sm;
        mrow[j] = mnew;
        o[0][j] *= alpha; o[1][j] *= alpha; o[2][j] *= alpha; o[3][j] *= alpha;
        pb[(j + 8 * hi) * 32 + n]      = bf16r(pa);
        pb[(j + 8 * hi) * 32 + 16 + n] = bf16r(pc);
      }
      asm volatile("s_wait_dscnt 0" ::: "memory");  // P writes before reload

      v16bf pf = load_fragA(pb, 32, 0, lane);       // P as A operand (16x32)
#pragma unroll
      for (int nt = 0; nt < 4; nt++) {
        v16bf vf = load_fragB(vb + nt * 16 * 32, 32, 0, lane);
        o[nt] = wmma_bf16(pf, vf, o[nt]);
      }
    }
    __syncthreads();                             // all reads of buf[cur] done
  }

  // normalize, stage, store to O[b, s, h*64+d]
  unsigned short* tt = &ost[wave][0];
#pragma unroll
  for (int nt = 0; nt < 4; nt++)
#pragma unroll
    for (int j = 0; j < 8; j++)
      tt[(j + 8 * hi) * 64 + nt * 16 + n] = bf16r(o[nt][j] / lrow[j]);
  asm volatile("s_wait_dscnt 0" ::: "memory");
  int b = bh >> 4, h = bh & 15;
  unsigned short* out = O + ((size_t)b * Tc + qs + n) * (Hc * Dc) + h * Dc + hi * 32;
  const unsigned short* src = tt + n * 64 + hi * 32;
#pragma unroll
  for (int c = 0; c < 4; c++) ((v4u*)out)[c] = ((const v4u*)src)[c];
}

// ------------------------------ output GEMM --------------------------------
// y[B*T,1024] = O[B*T,1024](bf16) @ Wc (via WcT[f][c] bf16), f32 out.
__global__ void __launch_bounds__(256) out_gemm(
    const unsigned short* __restrict__ Ob, const unsigned short* __restrict__ WcT,
    float* __restrict__ y) {
  __shared__ float fst[8][16 * 64];
  int wave = threadIdx.x >> 5, lane = threadIdx.x & 31;
  int wid = blockIdx.x * 8 + wave;
  int ft = wid & 15;
  int rblk = wid >> 4;
  int n = lane & 15, hi = lane >> 4;
  const unsigned short* Ab = Ob + (size_t)rblk * 16 * Ec;
  const unsigned short* Bb = WcT + (size_t)ft * 64 * Ec;

  v8f z = {0.f, 0.f, 0.f, 0.f, 0.f, 0.f, 0.f, 0.f};
  v8f acc[4];
#pragma unroll
  for (int i = 0; i < 4; i++) acc[i] = z;

  v16bf a  = load_fragA(Ab, Ec, 0, lane);
  v16bf b0 = load_fragB(Bb + 0 * 16 * Ec, Ec, 0, lane);
  v16bf b1 = load_fragB(Bb + 1 * 16 * Ec, Ec, 0, lane);
  v16bf b2 = load_fragB(Bb + 2 * 16 * Ec, Ec, 0, lane);
  v16bf b3 = load_fragB(Bb + 3 * 16 * Ec, Ec, 0, lane);
  for (int c0 = 0; c0 < Ec - 32; c0 += 32) {
    v16bf an = load_fragA(Ab, Ec, c0 + 32, lane);
    v16bf c0f = load_fragB(Bb + 0 * 16 * Ec, Ec, c0 + 32, lane);
    v16bf c1f = load_fragB(Bb + 1 * 16 * Ec, Ec, c0 + 32, lane);
    v16bf c2f = load_fragB(Bb + 2 * 16 * Ec, Ec, c0 + 32, lane);
    v16bf c3f = load_fragB(Bb + 3 * 16 * Ec, Ec, c0 + 32, lane);
    acc[0] = wmma_bf16(a, b0, acc[0]);
    acc[1] = wmma_bf16(a, b1, acc[1]);
    acc[2] = wmma_bf16(a, b2, acc[2]);
    acc[3] = wmma_bf16(a, b3, acc[3]);
    a = an; b0 = c0f; b1 = c1f; b2 = c2f; b3 = c3f;
  }
  acc[0] = wmma_bf16(a, b0, acc[0]);
  acc[1] = wmma_bf16(a, b1, acc[1]);
  acc[2] = wmma_bf16(a, b2, acc[2]);
  acc[3] = wmma_bf16(a, b3, acc[3]);

  float* tt = &fst[wave][0];
#pragma unroll
  for (int nt = 0; nt < 4; nt++)
#pragma unroll
    for (int j = 0; j < 8; j++)
      tt[(j + 8 * hi) * 64 + nt * 16 + n] = acc[nt][j];
  asm volatile("s_wait_dscnt 0" ::: "memory");
  float* out = y + ((size_t)rblk * 16 + n) * Ec + ft * 64 + hi * 32;
  const float* src = tt + n * 64 + hi * 32;
#pragma unroll
  for (int c = 0; c < 8; c++) ((v4u*)out)[c] = ((const v4u*)src)[c];
}

// ------------------------------- launcher ----------------------------------

extern "C" void kernel_launch(void* const* d_in, const int* in_sizes, int n_in,
                              void* d_out, int out_size, void* d_ws, size_t ws_size,
                              hipStream_t stream) {
  (void)in_sizes; (void)n_in; (void)out_size; (void)ws_size;
  const float* x  = (const float*)d_in[0];
  const float* Wq = (const float*)d_in[1];
  const float* Wk = (const float*)d_in[2];
  const float* Wv = (const float*)d_in[3];
  const float* Wc = (const float*)d_in[4];

  char* w = (char*)d_ws;
  size_t off = 0;
  auto take = [&](size_t bytes) {
    void* p = w + off;
    off += (bytes + 255) & ~(size_t)255;
    return p;
  };
  unsigned short* xb  = (unsigned short*)take((size_t)Bc * Tc * Ec * 2);      // 16 MB
  unsigned short* WqT = (unsigned short*)take((size_t)Hc * Dc * Ec * 2);      //  2 MB
  unsigned short* WkT = (unsigned short*)take((size_t)Hc * Dc * Ec * 2);
  unsigned short* WvT = (unsigned short*)take((size_t)Hc * Dc * Ec * 2);
  unsigned short* WcT = (unsigned short*)take((size_t)Hc * Dc * Ec * 2);
  unsigned short* Qb  = (unsigned short*)take((size_t)Bc * Hc * Tc * Dc * 2); // 16 MB
  unsigned short* Kb  = (unsigned short*)take((size_t)Bc * Hc * Tc * Dc * 2);
  unsigned short* Vtb = (unsigned short*)take((size_t)Bc * Hc * Tc * Dc * 2);
  unsigned short* Ob  = (unsigned short*)take((size_t)Bc * Tc * Hc * Dc * 2);

  cvt_bf16<<<2048, 256, 0, stream>>>(x, xb, (long)Bc * Tc * Ec);
  transpose_bf16<<<1024, 256, 0, stream>>>(Wq, WqT, Hc, Ec, Dc);
  transpose_bf16<<<1024, 256, 0, stream>>>(Wk, WkT, Hc, Ec, Dc);
  transpose_bf16<<<1024, 256, 0, stream>>>(Wv, WvT, Hc, Ec, Dc);
  transpose_bf16<<<1024, 256, 0, stream>>>(Wc, WcT, 1, Hc * Dc, Ec);

  // 3 mats * 4 b * 16 h * 128 sblk = 24576 waves / 8 per block
  qkv_gemm<<<3072, 256, 0, stream>>>(xb, WqT, WkT, WvT, Qb, Kb, Vtb);
  // 4*16*128 = 8192 waves
  attn_kernel<<<1024, 256, 0, stream>>>(Qb, Kb, Vtb, Ob);
  // 512 rblk * 16 ftile = 8192 waves
  out_gemm<<<1024, 256, 0, stream>>>(Ob, WcT, (float*)d_out);
}
